// PreJointNetFace_50414326121240
// MI455X (gfx1250) — matrix operand order repaired and checked by reading, not compiled, not run on hardware
//
#include <hip/hip_runtime.h>
#include <hip/hip_bf16.h>

typedef __attribute__((ext_vector_type(16))) __bf16 v16bf;
typedef __attribute__((ext_vector_type(8)))  float  v8f;

#define KCH_G1 22   // ceil(700/32) K-chunks for grid GEMM1
#define KCH_H   4   // 128/32      K-chunks for hidden GEMMs

// ---- CDNA5 async global->LDS path (ASYNCcnt), with safe fallback ----------
#if defined(__has_builtin)
#if __has_builtin(__builtin_amdgcn_global_load_async_to_lds_b128) && \
    __has_builtin(__builtin_amdgcn_s_wait_asynccnt)
#define USE_ASYNC_LDS 1
#endif
#endif
#ifndef USE_ASYNC_LDS
#define USE_ASYNC_LDS 0
#endif

#if USE_ASYNC_LDS
typedef int v4i_g __attribute__((vector_size(16)));               // GCC-style int4
typedef __attribute__((address_space(1))) v4i_g* g_v4i_ptr;        // global int4*
typedef __attribute__((address_space(3))) v4i_g* l_v4i_ptr;        // LDS int4*
#endif

static __device__ __forceinline__ void async_cp16(float* lds_dst, const float* gsrc) {
#if USE_ASYNC_LDS
  __builtin_amdgcn_global_load_async_to_lds_b128(
      (g_v4i_ptr)(uintptr_t)gsrc,
      (l_v4i_ptr)(unsigned int)(uintptr_t)lds_dst, 0, 0);
#else
  *(float4*)lds_dst = *(const float4*)gsrc;
#endif
}
static __device__ __forceinline__ void async_wait0() {
#if USE_ASYNC_LDS
  __builtin_amdgcn_s_wait_asynccnt(0);
#endif
}

static __device__ __forceinline__ __bf16 f2bf(float f) { return (__bf16)f; }

static __device__ __forceinline__ v8f wmma_bf16(v16bf a, v16bf b, v8f c) {
  return __builtin_amdgcn_wmma_f32_16x16x32_bf16(
      /*neg_a=*/false, a, /*neg_b=*/false, b,
      /*c_mod=*/(short)0, c, /*reuse_a=*/false, /*reuse_b=*/false);
}

// Convert f32 weight W[K x 128] (row-major, zero-padded K -> kchunks*32) into
// WMMA B-fragment order (B 32x16 bf16: lanes hold columns; lanes 0-15 K=0..7,16..23,
// lanes 16-31 K=8..15,24..31):
//   dst[((ct*kchunks + kc)*32 + lane)*16 + e]
//     = bf16( W[ kc*32 + (e>>3)*16 + (lane>>4)*8 + (e&7) ][ ct*16 + (lane&15) ] )
__global__ __launch_bounds__(256) void conv_weight_kernel(
    const float* __restrict__ W, void* __restrict__ dst_v,
    int Kvalid, int kchunks) {
  __bf16* dst = (__bf16*)dst_v;
  int t = blockIdx.x * 256 + threadIdx.x;
  int total = 8 * kchunks * 32;
  if (t >= total) return;
  int lane = t & 31;
  int kc   = (t >> 5) % kchunks;
  int ct   = t / (32 * kchunks);
  v16bf frag;
#pragma unroll
  for (int e = 0; e < 16; ++e) {
    int k = kc * 32 + ((e >> 3) << 4) + ((lane >> 4) << 3) + (e & 7);
    int n = ct * 16 + (lane & 15);
    float v = (k < Kvalid) ? W[k * 128 + n] : 0.0f;
    frag[e] = f2bf(v);
  }
  *(v16bf*)(dst + (size_t)((ct * kchunks + kc) * 32 + lane) * 16) = frag;
}

// One graph: out[r, 0:128]   = mask * (ELU(grid[r] @ Wg1 + bg1) @ Wg2 + bg2)
//            out[r, 128:256] = mask * (ELU(ent[r]  @ We1 + be1) @ We2 + be2)
//
// Block = 256 threads = 8 wave32. Wave w owns rows [blk*128 + w*16, +16) and
// computes all 128 output columns (8 ctiles). ALL LDS use is wave-private
// (A stage, H tile, ent stage), so the kernel is completely barrier-free;
// A staging is async-to-LDS with per-wave ping-pong double buffering, and all
// 8 B fragments of a chunk are register-blocked so their loads issue as one
// clause ahead of the 8-WMMA burst.
//
// Per-wave LDS slab (1280 floats = 5120 B):
//   [0    .. 1152) : A stage, 2 buffers x (16 rows x 36 f32)
//   [0    .. 1088) : H tile reuse, 16 rows x 136 bf16 (time-shared with A)
//   [1088 .. 1280) : ent stage, 16 rows x 12 f32
__global__ __launch_bounds__(256) void prejoint_fused_kernel(
    const float* __restrict__ grid, const float* __restrict__ ent,
    const int* __restrict__ isf,
    const void* __restrict__ Wg1v, const float* __restrict__ bg1,
    const void* __restrict__ Wg2v, const float* __restrict__ bg2,
    const void* __restrict__ We1v, const float* __restrict__ be1,
    const void* __restrict__ We2v, const float* __restrict__ be2,
    float* __restrict__ out, int N) {
  const __bf16* Wg1b = (const __bf16*)Wg1v;
  const __bf16* Wg2b = (const __bf16*)Wg2v;
  const __bf16* We1b = (const __bf16*)We1v;
  const __bf16* We2b = (const __bf16*)We2v;

  __shared__ __attribute__((aligned(16))) float ldsPool[8 * 1280];  // 40 KB

  const int tid  = threadIdx.x;
  const int wave = tid >> 5;
  const int lane = tid & 31;
  const long long rowBase = (long long)blockIdx.x * 128 + wave * 16;

  float*  Abase = ldsPool + wave * 1280;
  __bf16* Hbase = (__bf16*)Abase;
  float*  Ebase = Abase + 1088;

  const int col  = lane & 15;            // A-frag local row / C column-in-tile
  const int koff = (lane >> 4) << 3;     // hi half-lanes: K += 8 (A layout)
  const int srow = lane >> 1;            // staging: local row 0..15
  const int shalf = (lane & 1) << 4;     // staging: K half 0/16

  // per-row is_face mask (C layout: VGPR v -> local row v + 8*(lane>>4))
  const long long r0 = rowBase + ((lane >> 4) << 3);
  float mrow[8];
#pragma unroll
  for (int v = 0; v < 8; ++v) {
    long long rr = r0 + v;
    mrow[v] = (rr < N && isf[rr] > 0) ? 1.0f : 0.0f;
  }

  const v8f zero8 = {0.f, 0.f, 0.f, 0.f, 0.f, 0.f, 0.f, 0.f};

  // stage one 16x32 f32 chunk of this wave's rows into A buffer `buf`
  auto stageA = [&](int c, int buf) {
    long long grow = rowBase + srow;
    float* dst = Abase + buf * 576 + srow * 36 + shalf;
    if (grow < N) {
      const float* gp = grid + grow * 700 + c * 32 + shalf;
#pragma unroll
      for (int q = 0; q < 4; ++q) {
        int k = c * 32 + shalf + q * 4;
        if (k < 700) async_cp16(dst + q * 4, gp + q * 4);
        else *(float4*)(dst + q * 4) = make_float4(0.f, 0.f, 0.f, 0.f);
      }
    } else {
#pragma unroll
      for (int q = 0; q < 4; ++q)
        *(float4*)(dst + q * 4) = make_float4(0.f, 0.f, 0.f, 0.f);
    }
  };

  // ---------------- grid path GEMM1: [16 x 700] @ [700 x 128] -------------
  v8f acc[8];
#pragma unroll
  for (int j = 0; j < 8; ++j) acc[j] = zero8;

  stageA(0, 0);
  for (int c = 0; c < KCH_G1; ++c) {
    const int cur = c & 1;

    // register-block all 8 B fragments: one big load clause
    v16bf bfr[8];
#pragma unroll
    for (int j = 0; j < 8; ++j)
      bfr[j] = *(const v16bf*)(Wg1b + (size_t)((j * KCH_G1 + c) * 32 + lane) * 16);

    async_wait0();                         // chunk c landed in buf `cur`
    if (c + 1 < KCH_G1) stageA(c + 1, 1 - cur);  // prefetch next (overlaps WMMA)

    const float* Ab = Abase + cur * 576;
    v16bf a;
#pragma unroll
    for (int e = 0; e < 16; ++e) {
      int k = ((e >> 3) << 4) + koff + (e & 7);
      a[e] = f2bf(Ab[col * 36 + k]);
    }
#pragma unroll
    for (int j = 0; j < 8; ++j) acc[j] = wmma_bf16(a, bfr[j], acc[j]);
  }

  // bias + ELU -> bf16 hidden tile (wave-private, in-order LDS reuse of A)
  const int rb = (lane >> 4) << 3;
#pragma unroll
  for (int j = 0; j < 8; ++j) {
    float bc = bg1[j * 16 + col];
#pragma unroll
    for (int v = 0; v < 8; ++v) {
      float x = acc[j][v] + bc;
      x = x > 0.f ? x : (__expf(x) - 1.0f);
      Hbase[(rb + v) * 136 + j * 16 + col] = f2bf(x);
    }
  }

  // ---------------- grid path GEMM2: [16 x 128] @ [128 x 128] -------------
  v8f acc2[8];
#pragma unroll
  for (int j = 0; j < 8; ++j) acc2[j] = zero8;
#pragma unroll
  for (int c = 0; c < KCH_H; ++c) {
    v16bf bfr[8];
#pragma unroll
    for (int j = 0; j < 8; ++j)
      bfr[j] = *(const v16bf*)(Wg2b + (size_t)((j * KCH_H + c) * 32 + lane) * 16);
    v16bf a;
#pragma unroll
    for (int e = 0; e < 16; ++e) {
      int k = c * 32 + ((e >> 3) << 4) + koff + (e & 7);
      a[e] = Hbase[col * 136 + k];
    }
#pragma unroll
    for (int j = 0; j < 8; ++j) acc2[j] = wmma_bf16(a, bfr[j], acc2[j]);
  }
  // bias + mask, store columns [0,128)
#pragma unroll
  for (int j = 0; j < 8; ++j) {
    float bc = bg2[j * 16 + col];
#pragma unroll
    for (int v = 0; v < 8; ++v) {
      long long rr = r0 + v;
      if (rr < N) out[rr * 256 + j * 16 + col] = (acc2[j][v] + bc) * mrow[v];
    }
  }

  // ---------------- ent path GEMM1: [16 x 9(->32)] @ [9 x 128] ------------
  for (int i = lane; i < 16 * 9; i += 32) {
    int r = i / 9, k = i - r * 9;
    long long rr = rowBase + r;
    Ebase[r * 12 + k] = (rr < N) ? ent[rr * 9 + k] : 0.f;
  }
  {
    v16bf bfr[8];
#pragma unroll
    for (int j = 0; j < 8; ++j)
      bfr[j] = *(const v16bf*)(We1b + (size_t)(j * 32 + lane) * 16);
    v16bf a;
#pragma unroll
    for (int e = 0; e < 16; ++e) {
      int k = ((e >> 3) << 4) + koff + (e & 7);
      float v = (k < 9) ? Ebase[col * 12 + k] : 0.f;
      a[e] = f2bf(v);
    }
#pragma unroll
    for (int j = 0; j < 8; ++j) acc[j] = zero8;
#pragma unroll
    for (int j = 0; j < 8; ++j) acc[j] = wmma_bf16(a, bfr[j], acc[j]);
  }
  // bias + ELU -> H (overwrites grid H, already consumed; in-order per wave)
#pragma unroll
  for (int j = 0; j < 8; ++j) {
    float bc = be1[j * 16 + col];
#pragma unroll
    for (int v = 0; v < 8; ++v) {
      float x = acc[j][v] + bc;
      x = x > 0.f ? x : (__expf(x) - 1.0f);
      Hbase[(rb + v) * 136 + j * 16 + col] = f2bf(x);
    }
  }

  // ---------------- ent path GEMM2: [16 x 128] @ [128 x 128] --------------
#pragma unroll
  for (int j = 0; j < 8; ++j) acc2[j] = zero8;
#pragma unroll
  for (int c = 0; c < KCH_H; ++c) {
    v16bf bfr[8];
#pragma unroll
    for (int j = 0; j < 8; ++j)
      bfr[j] = *(const v16bf*)(We2b + (size_t)((j * KCH_H + c) * 32 + lane) * 16);
    v16bf a;
#pragma unroll
    for (int e = 0; e < 16; ++e) {
      int k = c * 32 + ((e >> 3) << 4) + koff + (e & 7);
      a[e] = Hbase[col * 136 + k];
    }
#pragma unroll
    for (int j = 0; j < 8; ++j) acc2[j] = wmma_bf16(a, bfr[j], acc2[j]);
  }
  // bias + mask, store columns [128,256)
#pragma unroll
  for (int j = 0; j < 8; ++j) {
    float bc = be2[j * 16 + col];
#pragma unroll
    for (int v = 0; v < 8; ++v) {
      long long rr = r0 + v;
      if (rr < N) out[rr * 256 + 128 + j * 16 + col] = (acc2[j][v] + bc) * mrow[v];
    }
  }
}

extern "C" void kernel_launch(void* const* d_in, const int* in_sizes, int n_in,
                              void* d_out, int out_size, void* d_ws, size_t ws_size,
                              hipStream_t stream) {
  const int N = in_sizes[0] / 700;

  const float* grid1 = (const float*)d_in[0];
  const float* ent1  = (const float*)d_in[1];
  const int*   isf1  = (const int*)d_in[2];
  const float* grid2 = (const float*)d_in[3];
  const float* ent2  = (const float*)d_in[4];
  const int*   isf2  = (const int*)d_in[5];
  const float* Wg1 = (const float*)d_in[6];
  const float* bg1 = (const float*)d_in[7];
  const float* Wg2 = (const float*)d_in[8];
  const float* bg2 = (const float*)d_in[9];
  const float* We1 = (const float*)d_in[10];
  const float* be1 = (const float*)d_in[11];
  const float* We2 = (const float*)d_in[12];
  const float* be2 = (const float*)d_in[13];

  // d_ws layout (bf16 fragment buffers, 32B aligned offsets)
  char* ws = (char*)d_ws;
  void* Wg1b = (void*)(ws + 0);        // 8*22*32*16*2 = 180224 B
  void* Wg2b = (void*)(ws + 180224);   // 8* 4*32*16*2 =  32768 B
  void* We1b = (void*)(ws + 212992);   // 8* 1*32*16*2 =   8192 B
  void* We2b = (void*)(ws + 221184);   // 8* 4*32*16*2 =  32768 B

  conv_weight_kernel<<<(8 * KCH_G1 * 32 + 255) / 256, 256, 0, stream>>>(Wg1, Wg1b, 700, KCH_G1);
  conv_weight_kernel<<<(8 * KCH_H  * 32 + 255) / 256, 256, 0, stream>>>(Wg2, Wg2b, 128, KCH_H);
  conv_weight_kernel<<<(8 * 1      * 32 + 255) / 256, 256, 0, stream>>>(We1, We1b, 9, 1);
  conv_weight_kernel<<<(8 * KCH_H  * 32 + 255) / 256, 256, 0, stream>>>(We2, We2b, 128, KCH_H);

  const int nb = (N + 127) / 128;
  float* out1 = (float*)d_out;
  float* out2 = out1 + (size_t)N * 256;
  prejoint_fused_kernel<<<nb, 256, 0, stream>>>(grid1, ent1, isf1,
      Wg1b, bg1, Wg2b, bg2, We1b, be1, We2b, be2, out1, N);
  prejoint_fused_kernel<<<nb, 256, 0, stream>>>(grid2, ent2, isf2,
      Wg1b, bg1, Wg2b, bg2, We1b, be1, We2b, be2, out2, N);
}
